// SimilarityEnhancementModule_68813966016741
// MI455X (gfx1250) — matrix operand order repaired and checked by reading, not compiled, not run on hardware
//
#include <hip/hip_runtime.h>
#include <hip/hip_bf16.h>
#include <stdint.h>

// CDNA5 / gfx1250 fused similarity-enhancement kernel.
// d_ws: [midn f16 (B*N*D)] [final^T f16 (B*D*N)].
// Flash-attention-style single pass; softmax needs no rescaling since
// cosine sims are bounded by 1, so one streaming pass suffices.

typedef __attribute__((ext_vector_type(16))) _Float16 v16h;
typedef __attribute__((ext_vector_type(8)))  _Float16 v8h;
typedef __attribute__((ext_vector_type(8)))  float    v8f;

#define B_   8
#define N_   2048
#define D_   768
#define ROWS_PER_BLK 32
#define CB   64        // similarity columns per main-loop iteration
#define DCHUNK 192     // D columns owned by each wave in stage 2
#define NT   12        // DCHUNK / 16 accumulator tiles per wave
#define KT   (D_ / 32) // 24 k-steps in stage 1

// ---------------------------------------------------------------------------
// Kernel 1: L2-normalize mid features along D, emit f16 (one wave per row).
// ---------------------------------------------------------------------------
__global__ __launch_bounds__(256)
void sem_normalize_f16(const float* __restrict__ mid, _Float16* __restrict__ midn) {
    const int wave = threadIdx.x >> 5;
    const int lane = threadIdx.x & 31;
    const int row  = blockIdx.x * 8 + wave;   // B_*N_ rows total
    const float* src = mid  + (size_t)row * D_;
    _Float16*    dst = midn + (size_t)row * D_;

    float s = 0.f;
    for (int j = lane; j < D_; j += 32) { float v = src[j]; s += v * v; }
    #pragma unroll
    for (int m = 16; m >= 1; m >>= 1) s += __shfl_xor(s, m, 32);
    const float inv = 1.0f / fmaxf(sqrtf(s), 1e-12f);
    for (int j = lane; j < D_; j += 32) dst[j] = (_Float16)(src[j] * inv);
}

// ---------------------------------------------------------------------------
// Kernel 2: transpose+convert final [b][n][d] f32 -> finh_t [b][d][n] f16.
// Makes the V-block LDS staging a contiguous byte copy (async-DMA friendly).
// ---------------------------------------------------------------------------
__global__ __launch_bounds__(256)
void sem_transpose_f16(const float* __restrict__ finalf, _Float16* __restrict__ finh_t) {
    __shared__ _Float16 tile[32][33];
    const int b  = blockIdx.z;
    const int n0 = blockIdx.x * 32;
    const int d0 = blockIdx.y * 32;
    const int c  = threadIdx.x & 31;
    const int r0 = threadIdx.x >> 5;   // 0..7

    const float* src = finalf + ((size_t)b * N_ + n0) * D_ + d0;
    #pragma unroll
    for (int rr = 0; rr < 32; rr += 8)
        tile[r0 + rr][c] = (_Float16)src[(size_t)(r0 + rr) * D_ + c];
    __syncthreads();
    _Float16* dst = finh_t + ((size_t)b * D_ + d0) * N_ + n0;
    #pragma unroll
    for (int rr = 0; rr < 32; rr += 8)
        dst[(size_t)(r0 + rr) * N_ + c] = tile[c][r0 + rr];
}

// ---------------------------------------------------------------------------
// Kernel 3: fused  sim -> mask diag -> conf/softmax -> P@V -> calibrate.
// Grid: B_ * (N_/32) workgroups of 256 threads (8 wave32 waves).
//   wave = 4*g + wg : g = row-group (16 rows), wg = col-tile id (stage 1)
//                                              and D-chunk id (stage 2).
// Per column block: async-DMA V into LDS overlapped with the QK^T WMMA chain.
// ---------------------------------------------------------------------------
__global__ __launch_bounds__(256)
void sem_fused(const _Float16* __restrict__ midn,
               const _Float16* __restrict__ finh_t,
               const float*    __restrict__ finalf,
               float*          __restrict__ out) {
    extern __shared__ _Float16 smem[];
    _Float16* vlds = smem;                       // V block [d][r] f16  (96 KB)
    _Float16* plds = smem + D_ * CB;             // exp(S) [2][16][CB]   (4 KB)
    float*    red  = (float*)(smem + D_ * CB + 2 * 16 * CB); // [2][4][16][2]

    const int tid  = threadIdx.x;
    const int lane = tid & 31;
    const int wave = tid >> 5;
    const int g    = wave >> 2;
    const int wg   = wave & 3;
    const int half = lane >> 4;      // 0: lanes 0-15, 1: lanes 16-31
    const int ln   = lane & 15;

    const int b     = blockIdx.x / (N_ / ROWS_PER_BLK);
    const int rblk  = blockIdx.x % (N_ / ROWS_PER_BLK);
    const int rbase = rblk * ROWS_PER_BLK;
    const int rg    = rbase + 16 * g;            // this row-group's 16 rows
    const int dbase = wg * DCHUNK;

    const size_t batchOff = (size_t)b * N_ * D_;
    const _Float16* midb  = midn   + batchOff;
    const _Float16* vbase = finh_t + (size_t)b * D_ * N_;   // [d][n]
    const float*    finb  = finalf + batchOff;

    v8f acc[NT];
    #pragma unroll
    for (int t = 0; t < NT; ++t) acc[t] = v8f{};
    v8f sum_p = v8f{};
    v8f conf_p;
    #pragma unroll
    for (int i = 0; i < 8; ++i) conf_p[i] = -1e30f;

    union U { v16h v; v8h h[2]; };

    for (int cb = 0; cb < N_; cb += CB) {
        __syncthreads();   // all readers of vlds/plds from prev iter are done

        // ---- async-DMA V block: vlds[d*CB + r] = finh_t[b][d][cb + r] ----
        // Contiguous 16B per lane; overlaps with stage-1 WMMA below.
        {
            const _Float16* vsrc = vbase + cb;
            for (int idx = tid; idx < (D_ * CB) / 8; idx += 256) {
                const int linear = idx * 8;
                const int d = linear / CB;
                const int r = linear % CB;
                const _Float16* gp = vsrc + (size_t)d * N_ + r;
                const uint32_t ldsoff = (uint32_t)(uintptr_t)(vlds + linear);
                asm volatile("global_load_async_to_lds_b128 %0, %1, off"
                             :: "v"(ldsoff), "v"((uint64_t)(uintptr_t)gp)
                             : "memory");
                if (cb + CB < N_) __builtin_prefetch(gp + CB, 0, 1);
            }
        }

        // ---- stage 1: S(16x16) = Q(rows rg..) x K(cols cb+16*wg..)^T ----
        // Double-buffered B operands so loads run ahead of the WMMA chain.
        v8f s = v8f{};
        const int ccol = cb + 16 * wg;
        const _Float16* arow  = midb + (size_t)(rg   + ln) * D_;
        const _Float16* bbase = midb + (size_t)(ccol + ln) * D_ + (half ? 8 : 0);
        U bm[2];
        bm[0].h[0] = *(const v8h*)(bbase);
        bm[0].h[1] = *(const v8h*)(bbase + 16);
        #pragma unroll
        for (int kt = 0; kt < KT; ++kt) {
            const int k = kt * 32;
            if (kt + 1 < KT)
            { bm[(kt + 1) & 1].h[0] = *(const v8h*)(bbase + k + 32);
              bm[(kt + 1) & 1].h[1] = *(const v8h*)(bbase + k + 48); }
            U a;
            const _Float16* ap = arow + k + (half ? 8 : 0);   // ISA A-layout
            a.h[0] = *(const v8h*)(ap);
            a.h[1] = *(const v8h*)(ap + 16);
            s = __builtin_amdgcn_wmma_f32_16x16x32_f16(false, a.v, false, bm[kt & 1].v,
                                                       (short)0, s, false, false);
        }

        // diag -> 0, track confidence + softmax sum, stash exp(S) in LDS
        {
            const int gcol = ccol + ln;
            #pragma unroll
            for (int i = 0; i < 8; ++i) {
                const int grow = rg + i + 8 * half;
                const float sv = (grow == gcol) ? 0.f : s[i];
                conf_p[i] = fmaxf(conf_p[i], sv);
                const float e = __expf(sv);      // TEMPERATURE == 1
                sum_p[i] += e;
                plds[g * (16 * CB) + (i + 8 * half) * CB + 16 * wg + ln] = (_Float16)e;
            }
        }

        asm volatile("s_wait_asynccnt 0x0" ::: "memory");  // V block landed
        __syncthreads();

        // ---- stage 2: acc += P(16 x CB) x V(CB x DCHUNK), LDS double-buffered ----
        #pragma unroll
        for (int kk = 0; kk < 2; ++kk) {
            U a;
            const _Float16* pp = plds + g * (16 * CB) + ln * CB + 32 * kk + (half ? 8 : 0);
            a.h[0] = *(const v8h*)(pp);
            a.h[1] = *(const v8h*)(pp + 16);
            const int roff = 32 * kk + 8 * half;
            const _Float16* vp0 = vlds + (size_t)(dbase + ln) * CB + roff;
            U bv[2];
            bv[0].h[0] = *(const v8h*)(vp0);
            bv[0].h[1] = *(const v8h*)(vp0 + 16);
            #pragma unroll
            for (int jt = 0; jt < NT; ++jt) {
                if (jt + 1 < NT) {
                    const _Float16* vp = vp0 + (size_t)(16 * (jt + 1)) * CB;
                    bv[(jt + 1) & 1].h[0] = *(const v8h*)(vp);
                    bv[(jt + 1) & 1].h[1] = *(const v8h*)(vp + 16);
                }
                acc[jt] = __builtin_amdgcn_wmma_f32_16x16x32_f16(false, a.v, false, bv[jt & 1].v,
                                                                 (short)0, acc[jt], false, false);
            }
        }
    }

    // ---- cross-lane reduction of per-row stats within 16-lane halves ----
    #pragma unroll
    for (int i = 0; i < 8; ++i) {
        float sm = sum_p[i], cf = conf_p[i];
        #pragma unroll
        for (int m = 8; m >= 1; m >>= 1) {
            sm += __shfl_xor(sm, m, 32);
            cf  = fmaxf(cf, __shfl_xor(cf, m, 32));
        }
        sum_p[i] = sm; conf_p[i] = cf;
    }
    if (ln == 0) {
        #pragma unroll
        for (int i = 0; i < 8; ++i) {
            const int row = i + 8 * half;
            red[((g * 4 + wg) * 16 + row) * 2 + 0] = sum_p[i];
            red[((g * 4 + wg) * 16 + row) * 2 + 1] = conf_p[i];
        }
    }
    __syncthreads();

    float rsum[8], rconf[8];
    #pragma unroll
    for (int i = 0; i < 8; ++i) {
        const int row = i + 8 * half;
        float sm = 0.f, cf = -1e30f;
        #pragma unroll
        for (int w = 0; w < 4; ++w) {
            sm += red[((g * 4 + w) * 16 + row) * 2 + 0];
            cf  = fmaxf(cf, red[((g * 4 + w) * 16 + row) * 2 + 1]);
        }
        rsum[i] = sm; rconf[i] = cf;
    }

    // ---- epilogue: out = c * (acc / rsum) + (1 - c) * final ----
    #pragma unroll
    for (int jt = 0; jt < NT; ++jt) {
        const int d = dbase + 16 * jt + ln;
        #pragma unroll
        for (int i = 0; i < 8; ++i) {
            const int grow = rg + i + 8 * half;
            const size_t off = (size_t)grow * D_ + d;
            const float fv = finb[off];
            const float refined = acc[jt][i] / rsum[i];
            const float c = rconf[i];
            out[batchOff + off] = c * refined + (1.f - c) * fv;
        }
    }
}

// ---------------------------------------------------------------------------
extern "C" void kernel_launch(void* const* d_in, const int* in_sizes, int n_in,
                              void* d_out, int out_size, void* d_ws, size_t ws_size,
                              hipStream_t stream) {
    const float* finalf = (const float*)d_in[0];
    const float* midf   = (const float*)d_in[1];
    float* out = (float*)d_out;

    _Float16* midn   = (_Float16*)d_ws;                       // B*N*D f16
    _Float16* finh_t = midn + (size_t)B_ * N_ * D_;           // B*D*N f16

    const size_t shmem = (size_t)(D_ * CB + 2 * 16 * CB) * sizeof(_Float16)
                       + (size_t)(2 * 4 * 16 * 2) * sizeof(float);  // 103424 B
    (void)hipFuncSetAttribute((const void*)sem_fused,
                              hipFuncAttributeMaxDynamicSharedMemorySize,
                              (int)shmem);

    sem_normalize_f16<<<(B_ * N_) / 8, 256, 0, stream>>>(midf, midn);
    sem_transpose_f16<<<dim3(N_ / 32, D_ / 32, B_), 256, 0, stream>>>(finalf, finh_t);
    sem_fused<<<B_ * (N_ / ROWS_PER_BLK), 256, shmem, stream>>>(midn, finh_t, finalf, out);
}
